// NCL_50766513438740
// MI455X (gfx1250) — matrix operand order repaired and checked by reading, not compiled, not run on hardware
//
#include <hip/hip_runtime.h>

typedef __attribute__((ext_vector_type(16))) _Float16 v16h;
typedef __attribute__((ext_vector_type(8)))  _Float16 v8h;
typedef __attribute__((ext_vector_type(2)))  _Float16 v2h;
typedef __attribute__((ext_vector_type(8)))  float    v8f;

#define TWO_PI_F   6.28318530717958647692f
// beta = 20 folded into exp2/log2 constants:
//   softplus(20z)/20 = max(z,0) + (ln2/20) * log2(1 + exp2(-20*log2e*|z|))
//   sigmoid(20z)     = rcp(1 + exp2(-20*log2e*z))
#define K_E2       28.853900817779268f    // 20 * log2(e)
#define C_LN       0.03465735902799726f   // ln(2) / 20

__device__ __forceinline__ float sigmoid20(float z) {
  float t = __builtin_amdgcn_exp2f(-K_E2 * z);
  return __builtin_amdgcn_rcpf(1.0f + t);
}
__device__ __forceinline__ float softplus20(float z) {
  float t = __builtin_amdgcn_exp2f(-K_E2 * fabsf(z));
  return fmaf(__builtin_amdgcn_logf(1.0f + t), C_LN, fmaxf(z, 0.0f));
}

__device__ __forceinline__ v16h cat16(v8h lo, v8h hi) {
  return __builtin_shufflevector(lo, hi, 0,1,2,3,4,5,6,7,8,9,10,11,12,13,14,15);
}

// B fragment for v_wmma_f32_16x16x32_f16.
// T is row-major [N=32][K=32] f16, i.e. row n of T = column n of the B matrix.
// Lanes 0-15 hold K=0..7 then K16..23 of column N=(lane&15)+16*nt; lanes 16-31 hold K8..15 then K24..31.
__device__ __forceinline__ v16h load_bfrag(const _Float16* T, int lane, int nt) {
  int n  = nt * 16 + (lane & 15);
  int kh = (lane >> 4) * 8;
  const _Float16* p = T + n * 32 + kh;
  v8h lo = *(const v8h*)(p);
  v8h hi = *(const v8h*)(p + 16);
  return cat16(lo, hi);
}

// A fragment: T row-major [M=32][K=32] f16.
// Lanes 0-15: row M=mt*16+(lane&15), halves = K0..7 then K16..23.
// Lanes 16-31: same rows, halves = K8..15 then K24..31.
__device__ __forceinline__ v16h load_afrag(const _Float16* T, int lane, int mt) {
  int m  = mt * 16 + (lane & 15);
  int ks = (lane >> 4) * 8;
  const _Float16* p = T + m * 32 + ks;
  v8h lo = *(const v8h*)(p);
  v8h hi = *(const v8h*)(p + 16);
  return cat16(lo, hi);
}

#define LDS_FENCE() asm volatile("s_wait_dscnt 0" ::: "memory")

__global__ __launch_bounds__(256) void fluid_mlp_grad_kernel(
    const float* __restrict__ x,
    const float* __restrict__ W0, const float* __restrict__ b0,
    const float* __restrict__ W1, const float* __restrict__ b1,
    const float* __restrict__ W2, const float* __restrict__ b2,
    const float* __restrict__ W3, const float* __restrict__ b3,
    float* __restrict__ out)
{
  // f16 weight tiles: WT* = W^T (forward B), Wf* = W (backward B, since B = W^T there)
  __shared__ __align__(16) _Float16 WT1[1024], Wf1[1024], WT2[1024], Wf2[1024];
  __shared__ __align__(16) float sW0[160], sW3[160];
  __shared__ __align__(16) float sb0[32], sb1[32], sb2[32], sb3[8];
  __shared__ __align__(16) _Float16 tH[8][1024];   // per-wave activation staging
  __shared__ __align__(16) _Float16 tG[8][1024];   // per-wave gradient staging

  const int tid  = threadIdx.x;
  const int lane = tid & 31;
  const int wave = tid >> 5;

  // ---------------- weight prep (once per block) ----------------
  #pragma unroll
  for (int it = 0; it < 4; ++it) {
    int i = tid + it * 256;          // i = k*32 + n
    int k = i >> 5, n = i & 31;
    _Float16 w1 = (_Float16)W1[i];
    WT1[n*32 + k] = w1;  Wf1[i] = w1;
    _Float16 w2 = (_Float16)W2[i];
    WT2[n*32 + k] = w2;  Wf2[i] = w2;
  }
  if (tid < 160) { sW0[tid] = W0[tid]; sW3[tid] = W3[tid]; }
  if (tid < 32)  { sb0[tid] = b0[tid]; sb1[tid] = b1[tid]; sb2[tid] = b2[tid]; }
  if (tid < 5)   { sb3[tid] = b3[tid]; }
  __syncthreads();

  _Float16* myH = tH[wave];
  _Float16* myG = tG[wave];

  const int p = (blockIdx.x * 8 + wave) * 32 + lane;
  const float* xv = x + (size_t)p * 3;
  float t  = xv[0];
  float a1r = TWO_PI_F * xv[1];
  float a2r = TWO_PI_F * xv[2];
  // raw HW trans ops: args in [0, 2pi), well inside v_sin/v_cos valid range
  float sx = __builtin_amdgcn_sinf(a1r), cx = __builtin_amdgcn_cosf(a1r);
  float sy = __builtin_amdgcn_sinf(a2r), cy = __builtin_amdgcn_cosf(a2r);
  // embed e = {t, cx, sx, cy, sy}

  // ---------------- layer 0: 5 -> 32, per-lane VALU ----------------
  v2h s0p[16];                                  // sigmoid(20*z0), packed f16x2
  #pragma unroll
  for (int j = 0; j < 32; j += 2) {
    float za = sb0[j]   + t*sW0[j]     + cx*sW0[32+j]   + sx*sW0[64+j]
                        + cy*sW0[96+j] + sy*sW0[128+j];
    float zb = sb0[j+1] + t*sW0[j+1]   + cx*sW0[33+j]   + sx*sW0[65+j]
                        + cy*sW0[97+j] + sy*sW0[129+j];
    v2h hv = { (_Float16)softplus20(za), (_Float16)softplus20(zb) };
    *(v2h*)(myH + lane*32 + j) = hv;            // h0 row-major [point][feat]
    v2h sv = { (_Float16)sigmoid20(za), (_Float16)sigmoid20(zb) };
    s0p[j >> 1] = sv;
  }
  LDS_FENCE();

  // ---------------- layer 1: z1 = h0 @ W1 + b1 (WMMA) ----------------
  v8f z1t[2][2];
  {
    v16h a0 = load_afrag(myH, lane, 0);
    v16h a1 = load_afrag(myH, lane, 1);
    #pragma unroll
    for (int nt = 0; nt < 2; ++nt) {
      v16h bf = load_bfrag(WT1, lane, nt);
      float bb = sb1[nt*16 + (lane & 15)];
      v8f c = {bb, bb, bb, bb, bb, bb, bb, bb};
      z1t[0][nt] = __builtin_amdgcn_wmma_f32_16x16x32_f16(false, a0, false, bf, (short)0, c, false, false);
      z1t[1][nt] = __builtin_amdgcn_wmma_f32_16x16x32_f16(false, a1, false, bf, (short)0, c, false, false);
    }
  }
  // elementwise in D-layout: h1 -> myH, keep s1 in regs (same D-layout is reused in backward)
  v2h s1p[2][2][4];
  #pragma unroll
  for (int mt = 0; mt < 2; ++mt)
    #pragma unroll
    for (int nt = 0; nt < 2; ++nt) {
      int ncol  = nt*16 + (lane & 15);
      int mbase = mt*16 + (lane >> 4) * 8;
      #pragma unroll
      for (int r = 0; r < 8; r += 2) {
        float z0v = z1t[mt][nt][r];
        float z1v = z1t[mt][nt][r+1];
        myH[(mbase+r  )*32 + ncol] = (_Float16)softplus20(z0v);
        myH[(mbase+r+1)*32 + ncol] = (_Float16)softplus20(z1v);
        v2h sv = { (_Float16)sigmoid20(z0v), (_Float16)sigmoid20(z1v) };
        s1p[mt][nt][r >> 1] = sv;
      }
    }
  LDS_FENCE();

  // ---------------- layer 2: z2 = h1 @ W2 + b2 (WMMA) ----------------
  v8f z2t[2][2];
  {
    v16h a0 = load_afrag(myH, lane, 0);
    v16h a1 = load_afrag(myH, lane, 1);
    #pragma unroll
    for (int nt = 0; nt < 2; ++nt) {
      v16h bf = load_bfrag(WT2, lane, nt);
      float bb = sb2[nt*16 + (lane & 15)];
      v8f c = {bb, bb, bb, bb, bb, bb, bb, bb};
      z2t[0][nt] = __builtin_amdgcn_wmma_f32_16x16x32_f16(false, a0, false, bf, (short)0, c, false, false);
      z2t[1][nt] = __builtin_amdgcn_wmma_f32_16x16x32_f16(false, a1, false, bf, (short)0, c, false, false);
    }
  }
  // elementwise: h2 -> myH; G2 = W3[:,4] * sigmoid(20*z2) -> myG (backward seed)
  #pragma unroll
  for (int mt = 0; mt < 2; ++mt)
    #pragma unroll
    for (int nt = 0; nt < 2; ++nt) {
      int ncol  = nt*16 + (lane & 15);
      int mbase = mt*16 + (lane >> 4) * 8;
      float w3c = sW3[ncol*5 + 4];
      #pragma unroll
      for (int r = 0; r < 8; ++r) {
        float zv = z2t[mt][nt][r];
        myH[(mbase+r)*32 + ncol] = (_Float16)softplus20(zv);
        myG[(mbase+r)*32 + ncol] = (_Float16)(w3c * sigmoid20(zv));
      }
    }
  LDS_FENCE();

  // ---------------- output head: y = h2 @ W3 + b3 (per-lane) ----------------
  {
    const v8h* rp = (const v8h*)(myH + lane*32);
    v8h hrc[4] = { rp[0], rp[1], rp[2], rp[3] };
    float y0 = sb3[0], y1 = sb3[1], y2 = sb3[2], y3 = sb3[3];
    #pragma unroll
    for (int c2 = 0; c2 < 4; ++c2)
      #pragma unroll
      for (int jj = 0; jj < 8; ++jj) {
        int j = c2*8 + jj;
        float hj = (float)hrc[c2][jj];
        y0 = fmaf(hj, sW3[j*5+0], y0);
        y1 = fmaf(hj, sW3[j*5+1], y1);
        y2 = fmaf(hj, sW3[j*5+2], y2);
        y3 = fmaf(hj, sW3[j*5+3], y3);
      }
    float* op = out + (size_t)p * 6;
    *(float2*)(op)     = make_float2(y0, y1);
    *(float2*)(op + 2) = make_float2(y2, y3);
  }

  // ---------------- backward: g1 = G2 @ W2^T (WMMA, B from Wf2) ----------------
  v8f g1t[2][2];
  {
    v16h a0 = load_afrag(myG, lane, 0);
    v16h a1 = load_afrag(myG, lane, 1);
    v8f zc = {0.f, 0.f, 0.f, 0.f, 0.f, 0.f, 0.f, 0.f};
    #pragma unroll
    for (int nt = 0; nt < 2; ++nt) {
      v16h bf = load_bfrag(Wf2, lane, nt);
      g1t[0][nt] = __builtin_amdgcn_wmma_f32_16x16x32_f16(false, a0, false, bf, (short)0, zc, false, false);
      g1t[1][nt] = __builtin_amdgcn_wmma_f32_16x16x32_f16(false, a1, false, bf, (short)0, zc, false, false);
    }
  }
  // dz1 = g1 * s1 (both in identical D-layout) -> myG
  #pragma unroll
  for (int mt = 0; mt < 2; ++mt)
    #pragma unroll
    for (int nt = 0; nt < 2; ++nt) {
      int ncol  = nt*16 + (lane & 15);
      int mbase = mt*16 + (lane >> 4) * 8;
      #pragma unroll
      for (int r = 0; r < 8; ++r) {
        float d = g1t[mt][nt][r] * (float)s1p[mt][nt][r >> 1][r & 1];
        myG[(mbase+r)*32 + ncol] = (_Float16)d;
      }
    }
  LDS_FENCE();

  // ---------------- backward: g0 = dz1 @ W1^T (WMMA, B from Wf1) ----------------
  v8f g0t[2][2];
  {
    v16h a0 = load_afrag(myG, lane, 0);
    v16h a1 = load_afrag(myG, lane, 1);
    v8f zc = {0.f, 0.f, 0.f, 0.f, 0.f, 0.f, 0.f, 0.f};
    #pragma unroll
    for (int nt = 0; nt < 2; ++nt) {
      v16h bf = load_bfrag(Wf1, lane, nt);
      g0t[0][nt] = __builtin_amdgcn_wmma_f32_16x16x32_f16(false, a0, false, bf, (short)0, zc, false, false);
      g0t[1][nt] = __builtin_amdgcn_wmma_f32_16x16x32_f16(false, a1, false, bf, (short)0, zc, false, false);
    }
  }
  // stage g0 -> myG so each lane can read its own point's row
  #pragma unroll
  for (int mt = 0; mt < 2; ++mt)
    #pragma unroll
    for (int nt = 0; nt < 2; ++nt) {
      int ncol  = nt*16 + (lane & 15);
      int mbase = mt*16 + (lane >> 4) * 8;
      #pragma unroll
      for (int r = 0; r < 8; ++r)
        myG[(mbase+r)*32 + ncol] = (_Float16)g0t[mt][nt][r];
    }
  LDS_FENCE();

  // ---------------- backward tail: dz0 = g0*s0; ge = W0 @ dz0; embed chain ----------------
  {
    const v8h* rp = (const v8h*)(myG + lane*32);
    v8h grc[4] = { rp[0], rp[1], rp[2], rp[3] };
    float ge1 = 0.f, ge2 = 0.f, ge3 = 0.f, ge4 = 0.f;
    #pragma unroll
    for (int c2 = 0; c2 < 4; ++c2)
      #pragma unroll
      for (int jj = 0; jj < 8; ++jj) {
        int j = c2*8 + jj;
        float dz = (float)grc[c2][jj] * (float)s0p[j >> 1][j & 1];
        ge1 = fmaf(dz, sW0[32 + j],  ge1);
        ge2 = fmaf(dz, sW0[64 + j],  ge2);
        ge3 = fmaf(dz, sW0[96 + j],  ge3);
        ge4 = fmaf(dz, sW0[128 + j], ge4);
      }
    float dx1 = TWO_PI_F * (ge2 * cx - ge1 * sx);
    float dx2 = TWO_PI_F * (ge4 * cy - ge3 * sy);
    float* op = out + (size_t)p * 6;
    *(float2*)(op + 4) = make_float2(dx2, -dx1);   // B = [ds2, -ds1]
  }
}

extern "C" void kernel_launch(void* const* d_in, const int* in_sizes, int n_in,
                              void* d_out, int out_size, void* d_ws, size_t ws_size,
                              hipStream_t stream) {
  const float* x  = (const float*)d_in[0];
  const float* W0 = (const float*)d_in[1];
  const float* b0 = (const float*)d_in[2];
  const float* W1 = (const float*)d_in[3];
  const float* b1 = (const float*)d_in[4];
  const float* W2 = (const float*)d_in[5];
  const float* b2 = (const float*)d_in[6];
  const float* W3 = (const float*)d_in[7];
  const float* b3 = (const float*)d_in[8];
  float* out = (float*)d_out;

  int N = in_sizes[0] / 3;          // 1048576
  int nblocks = N / 256;            // 256 points per block (8 waves x 32 points)
  fluid_mlp_grad_kernel<<<nblocks, 256, 0, stream>>>(x, W0, b0, W1, b1, W2, b2, W3, b3, out);
}
